// MolecularGATForRegression_30949534335546
// MI455X (gfx1250) — matrix operand order repaired and checked by reading, not compiled, not run on hardware
//
#include <hip/hip_runtime.h>
#include <hip/hip_bf16.h>

// ---------------------------------------------------------------------------
// MolecularGATForRegression on gfx1250 (MI455X).
// GEMMs: v_wmma_f32_16x16x32_bf16 with B staged in LDS via async-to-LDS DMA.
// Edge softmax/scatter: float4-vectorized atomics (the HBM-bound part).
// ---------------------------------------------------------------------------

#define NN      50000      // nodes
#define NE      800000     // original edges
#define ET      850000     // edges + self loops
#define NG      512        // graphs
#define HC      256        // per-node feature width after both GAT layers
#define NEGSL   0.2f
#define MT_PER_BLK 8       // M-tiles (16 rows each) per workgroup in GEMM

typedef __attribute__((ext_vector_type(16))) __bf16 v16bf;
typedef __attribute__((ext_vector_type(8)))  __bf16 v8bf;
typedef __attribute__((ext_vector_type(8)))  float  v8f;

// ------------------------------ utilities ----------------------------------

__global__ void fill_f32(float* p, float v, int n) {
    int i = blockIdx.x * blockDim.x + threadIdx.x;
    if (i < n) p[i] = v;
}

__device__ inline void atomicMaxF(float* addr, float val) {
    int* ia = (int*)addr;
    int old = __float_as_int(*addr);
    while (__int_as_float(old) < val) {
        int assumed = old;
        old = atomicCAS(ia, assumed, __float_as_int(val));
        if (old == assumed) break;
    }
}

__device__ inline float elu1(float v) { return v > 0.f ? v : (__expf(v) - 1.f); }

// --------------------------- operand packing -------------------------------

// x [NN,9] f32 -> bf16 padded to K=32
__global__ void pack_x(const float* __restrict__ x, __bf16* __restrict__ xb) {
    int i = blockIdx.x * blockDim.x + threadIdx.x;
    if (i >= NN * 32) return;
    int node = i >> 5, k = i & 31;
    xb[i] = (k < 9) ? (__bf16)x[node * 9 + k] : (__bf16)0.f;
}

// Pack weights W [K,N] (row-major f32) into WMMA B-fragment order, bf16,
// K zero-padded to Kp.  Layout: ((ntile*ksteps + kstep)*32 + lane)*16 + j
// where lane&15 = column-within-tile, lane>>4 selects K half,
// j = 0..15 covers K = kstep*32 + (lane>>4)*16 + j.
// Fragments of a 64-col strip (4 consecutive ntiles) are contiguous.
__global__ void pack_B(const float* __restrict__ W, __bf16* __restrict__ Bp,
                       int K, int Kp, int N) {
    int ksteps = Kp >> 5;
    int total  = (N >> 4) * ksteps * 32;
    int t = blockIdx.x * blockDim.x + threadIdx.x;
    if (t >= total) return;
    int lane  = t & 31;
    int kstep = (t >> 5) % ksteps;
    int ntile = (t >> 5) / ksteps;
    int ncol  = (ntile << 4) + (lane & 15);
    int kbase = (kstep << 5) + ((lane >> 4) << 4);
    __bf16* outp = Bp + (size_t)t * 16;
#pragma unroll
    for (int j = 0; j < 16; ++j) {
        int kk = kbase + j;
        outp[j] = (__bf16)((kk < K) ? W[(size_t)kk * N + ncol] : 0.f);
    }
}

// ------------------------------ WMMA GEMM ----------------------------------
// C[M,N] f32 = A[M,Kp] (bf16 row-major) @ Bp (fragment-packed bf16).
// One block: 8 waves = 8 consecutive 16-row M-tiles sharing one 64-col strip.
// The strip's B fragments (4*ksteps*512 bf16, <=32KB) are DMA'd to LDS with
// global_load_async_to_lds_b128 (ASYNCcnt), then WMMA reads B via ds_load.
__global__ __launch_bounds__(256)
void wmma_gemm_bf16(const __bf16* __restrict__ A,
                    const __bf16* __restrict__ Bp,
                    float* __restrict__ C,
                    int M, int N, int Kp) {
    __shared__ __bf16 lds_b[4 * 8 * 512];            // max ksteps=8 -> 32 KB

    int ksteps   = Kp >> 5;
    int strips_n = N >> 6;
    int nstrip   = blockIdx.x % strips_n;
    int mblk     = blockIdx.x / strips_n;

    int tid  = threadIdx.x;
    int lane = tid & 31;
    int wv   = tid >> 5;

    // ---- async-stage this strip's B fragments into LDS (16B per lane-op) ----
    {
        int bytes = 4 * ksteps * 512 * 2;
        const char* bsrc = (const char*)(Bp + (size_t)nstrip * (bytes / 2));
        unsigned lbase = (unsigned)(size_t)(&lds_b[0]);
        for (int off = tid * 16; off < bytes; off += 256 * 16) {
            unsigned    l = lbase + (unsigned)off;
            const char* g = bsrc + off;
            asm volatile("global_load_async_to_lds_b128 %0, %1, off"
                         :: "v"(l), "v"(g) : "memory");
        }
        asm volatile("s_wait_asynccnt 0x0" ::: "memory");
    }
    __syncthreads();

    int mtile = mblk * MT_PER_BLK + wv;
    if (mtile < (M >> 4)) {                           // wave-uniform: EXEC all-1
        int m0 = mtile << 4;
        v8f acc[4] = {v8f{}, v8f{}, v8f{}, v8f{}};

        int kh = lane >> 4;                           // K-half held by this lane
        const __bf16* abase = A + (size_t)(m0 + (lane & 15)) * Kp + kh * 16;
        const __bf16* lb    = &lds_b[(size_t)lane * 16];

        for (int ks = 0; ks < ksteps; ++ks) {
            // A fragment: contiguous K layout (v0-3: K base..base+7, v4-7: +8), matches B
            v8bf alo = *(const v8bf*)(abase + ks * 32);
            v8bf ahi = *(const v8bf*)(abase + ks * 32 + 8);
            v16bf a;
#pragma unroll
            for (int j = 0; j < 8; ++j) { a[j] = alo[j]; a[j + 8] = ahi[j]; }

#pragma unroll
            for (int t = 0; t < 4; ++t) {
                const __bf16* bp = lb + (size_t)(t * ksteps + ks) * 512;
                v8bf blo = *(const v8bf*)(bp);
                v8bf bhi = *(const v8bf*)(bp + 8);
                v16bf b;
#pragma unroll
                for (int j = 0; j < 8; ++j) { b[j] = blo[j]; b[j + 8] = bhi[j]; }
                acc[t] = __builtin_amdgcn_wmma_f32_16x16x32_bf16(
                    false, a, false, b, (short)0, acc[t], false, false);
            }
        }

        // C/D layout: lanes 0-15 rows m0..m0+7, lanes 16-31 rows m0+8..m0+15
        int cmb = m0 + kh * 8;
#pragma unroll
        for (int t = 0; t < 4; ++t) {
            int cn = (nstrip << 6) + (t << 4) + (lane & 15);
#pragma unroll
            for (int r = 0; r < 8; ++r)
                C[(size_t)(cmb + r) * N + cn] = acc[t][r];
        }
    }
}

// ------------------------- GAT attention pipeline --------------------------

// al_src/al_dst [NN,H]: per-(node,head) dot of h with attention vectors
__global__ void attn_scores(const float* __restrict__ h,
                            const float* __restrict__ a_src,
                            const float* __restrict__ a_dst,
                            float* als, float* ald, int H, int Cc) {
    int i = blockIdx.x * blockDim.x + threadIdx.x;
    if (i >= NN * H) return;
    int node = i / H, hh = i % H;
    const float* hp = h + (size_t)node * H * Cc + hh * Cc;
    const float* as = a_src + hh * Cc;
    const float* ad = a_dst + hh * Cc;
    float s0 = 0.f, s1 = 0.f;
    for (int c = 0; c < Cc; ++c) { float v = hp[c]; s0 += v * as[c]; s1 += v * ad[c]; }
    als[i] = s0; ald[i] = s1;
}

__device__ inline void edge_sd(const int* ei, int e, int& s, int& d) {
    if (e < NE) { s = ei[e]; d = ei[NE + e]; }
    else        { s = d = e - NE; }               // implicit self loops
}

__global__ void edge_max(const int* __restrict__ ei, const float* __restrict__ als,
                         const float* __restrict__ ald, float* emax, int H) {
    int i = blockIdx.x * blockDim.x + threadIdx.x;
    if (i >= ET * H) return;
    int e = i / H, hh = i % H, s, d;
    edge_sd(ei, e, s, d);
    float ev = als[s * H + hh] + ald[d * H + hh];
    ev = ev > 0.f ? ev : NEGSL * ev;
    atomicMaxF(&emax[d * H + hh], ev);
}

__global__ void edge_exp(const int* __restrict__ ei, const float* __restrict__ als,
                         const float* __restrict__ ald, const float* __restrict__ emax,
                         float* p, float* denom, int H) {
    int i = blockIdx.x * blockDim.x + threadIdx.x;
    if (i >= ET * H) return;
    int e = i / H, hh = i % H, s, d;
    edge_sd(ei, e, s, d);
    float ev = als[s * H + hh] + ald[d * H + hh];
    ev = ev > 0.f ? ev : NEGSL * ev;
    float pv = __expf(ev - emax[d * H + hh]);
    p[i] = pv;
    atomicAdd(&denom[d * H + hh], pv);
}

// agg[dst] += h[src] * alpha   (one thread per edge x 4 channels, float4 gather)
__global__ void aggregate4(const int* __restrict__ ei, const float* __restrict__ h,
                           const float* __restrict__ p, const float* __restrict__ denom,
                           float* agg, int H, int Cc) {
    int i = blockIdx.x * blockDim.x + threadIdx.x;   // ET * 64 = 54.4M
    if (i >= ET * (HC / 4)) return;
    int e = i / (HC / 4), q = i % (HC / 4);
    int c = q << 2, hh = c / Cc, s, d;
    edge_sd(ei, e, s, d);
    float alpha = p[e * H + hh] / (denom[d * H + hh] + 1e-16f);
    float4 hv = *(const float4*)(h + (size_t)s * HC + c);
    float* ap = agg + (size_t)d * HC + c;
    atomicAdd(ap + 0, hv.x * alpha);
    atomicAdd(ap + 1, hv.y * alpha);
    atomicAdd(ap + 2, hv.z * alpha);
    atomicAdd(ap + 3, hv.w * alpha);
}

// out_bf16 = bf16(elu(agg + bias)) : feeds next-layer WMMA A operand
__global__ void bias_elu_to_bf16(const float* __restrict__ agg,
                                 const float* __restrict__ bias,
                                 __bf16* __restrict__ out) {
    int i = blockIdx.x * blockDim.x + threadIdx.x;
    if (i >= NN * HC) return;
    out[i] = (__bf16)elu1(agg[i] + bias[i % HC]);
}

// ------------------------------- pooling -----------------------------------

// fused layer-2 bias+ELU with add/max pooling (+ node counts), float4
__global__ void pool_kernel4(const float* __restrict__ agg2, const float* __restrict__ b2,
                             const int* __restrict__ batch,
                             float* add_pool, float* max_pool, float* cnt) {
    int i = blockIdx.x * blockDim.x + threadIdx.x;   // NN * 64
    if (i >= NN * (HC / 4)) return;
    int node = i / (HC / 4), q = i % (HC / 4), c = q << 2;
    float4 av = *(const float4*)(agg2 + (size_t)node * HC + c);
    int g = batch[node];
    float vs[4] = {av.x, av.y, av.z, av.w};
    float* ap = add_pool + (size_t)g * HC + c;
    float* mp = max_pool + (size_t)g * HC + c;
#pragma unroll
    for (int j = 0; j < 4; ++j) {
        float v = elu1(vs[j] + b2[c + j]);
        atomicAdd(ap + j, v);
        atomicMaxF(mp + j, v);
    }
    if (q == 0) atomicAdd(&cnt[g], 1.0f);
}

// --------------------------------- MLP -------------------------------------

__global__ void mlp1_kernel(const float* __restrict__ add_pool,
                            const float* __restrict__ max_pool,
                            const float* __restrict__ cnt,
                            const float* __restrict__ w, const float* __restrict__ b,
                            float* gh) {
    int i = blockIdx.x * blockDim.x + threadIdx.x;   // NG*64
    if (i >= NG * 64) return;
    int g = i / 64, o = i % 64;
    float n = cnt[g], inv = 1.f / fmaxf(n, 1.f);
    const float* ap = add_pool + (size_t)g * HC;
    const float* mp = max_pool + (size_t)g * HC;
    float s = b[o];
    for (int j = 0; j < HC; ++j) {
        float a = ap[j];
        s += a * w[j * 64 + o];                       // add pool
        s += (a * inv) * w[(HC + j) * 64 + o];        // mean pool
        float m = (n > 0.f) ? mp[j] : 0.f;            // max pool (empty graph -> 0)
        s += m * w[(2 * HC + j) * 64 + o];
    }
    gh[i] = elu1(s);
}

__global__ void mlp2_kernel(const float* __restrict__ gh, const float* __restrict__ w2,
                            const float* __restrict__ b2, float* out) {
    int g = blockIdx.x * blockDim.x + threadIdx.x;
    if (g >= NG) return;
    float s = b2[0];
    for (int j = 0; j < 64; ++j) s += gh[g * 64 + j] * w2[j];
    out[g] = s;
}

// ------------------------------- launcher ----------------------------------

static inline int gdiv(long long n, int b) { return (int)((n + b - 1) / b); }

extern "C" void kernel_launch(void* const* d_in, const int* in_sizes, int n_in,
                              void* d_out, int out_size, void* d_ws, size_t ws_size,
                              hipStream_t stream) {
    (void)in_sizes; (void)n_in; (void)out_size; (void)ws_size;

    const float* x     = (const float*)d_in[0];
    const int*   ei    = (const int*)  d_in[1];
    const int*   batch = (const int*)  d_in[2];
    const float* W1    = (const float*)d_in[4];
    const float* as1   = (const float*)d_in[5];
    const float* ad1   = (const float*)d_in[6];
    const float* b1    = (const float*)d_in[7];
    const float* W2    = (const float*)d_in[8];
    const float* as2   = (const float*)d_in[9];
    const float* ad2   = (const float*)d_in[10];
    const float* b2    = (const float*)d_in[11];
    const float* l1w   = (const float*)d_in[12];
    const float* l1b   = (const float*)d_in[13];
    const float* l2w   = (const float*)d_in[14];
    const float* l2b   = (const float*)d_in[15];
    float* out = (float*)d_out;

    // ---- workspace carving (256B aligned, explicit reuse) ----
    char* base = (char*)d_ws;
    size_t off = 0;
    auto alloc = [&](size_t bytes) -> void* {
        void* p = base + off;
        off = (off + bytes + 255) & ~(size_t)255;
        return p;
    };
    __bf16* xb    = (__bf16*)alloc((size_t)NN * 32 * 2);
    __bf16* W1p   = (__bf16*)alloc((size_t)16 * 1 * 32 * 16 * 2);     // N/16 * Kp/32 * 32 * 16
    __bf16* W2p   = (__bf16*)alloc((size_t)16 * 8 * 32 * 16 * 2);
    float*  h1    = (float*) alloc((size_t)NN * HC * 4);              // reused as h2
    float*  als   = (float*) alloc((size_t)NN * 4 * 4);               // reused layer2 (H=2)
    float*  ald   = (float*) alloc((size_t)NN * 4 * 4);
    float*  emax  = (float*) alloc((size_t)NN * 4 * 4);
    float*  denom = (float*) alloc((size_t)NN * 4 * 4);
    float*  pbuf  = (float*) alloc((size_t)ET * 4 * 4);               // reused layer2
    float*  agg   = (float*) alloc((size_t)NN * HC * 4);              // reused layer2
    __bf16* h1b   = (__bf16*)alloc((size_t)NN * HC * 2);
    float*  addp  = (float*) alloc((size_t)NG * HC * 4);
    float*  maxp  = (float*) alloc((size_t)NG * HC * 4);
    float*  cnt   = (float*) alloc((size_t)NG * 4);
    float*  gh    = (float*) alloc((size_t)NG * 64 * 4);
    float*  h2    = h1;   // h1 (f32) dead after aggregate1/bias_elu1

    const int B = 256;
    const float NEG_INF = -3.4e38f;
    const int MTILES  = NN / 16;                       // 3125
    const int MBLKS   = (MTILES + MT_PER_BLK - 1) / MT_PER_BLK;
    const int STRIPS  = 256 / 64;                      // 4

    // ---- operand packing ----
    pack_x<<<gdiv((long long)NN * 32, B), B, 0, stream>>>(x, xb);
    pack_B<<<gdiv(16 * 1 * 32, B), B, 0, stream>>>(W1, W1p, 9, 32, 256);
    pack_B<<<gdiv(16 * 8 * 32, B), B, 0, stream>>>(W2, W2p, 256, 256, 256);

    // ---- layer 1: GEMM (WMMA bf16, LDS-staged B), softmax, scatter ----
    wmma_gemm_bf16<<<MBLKS * STRIPS, B, 0, stream>>>(xb, W1p, h1, NN, 256, 32);
    attn_scores<<<gdiv((long long)NN * 4, B), B, 0, stream>>>(h1, as1, ad1, als, ald, 4, 64);
    fill_f32<<<gdiv((long long)NN * 4, B), B, 0, stream>>>(emax, NEG_INF, NN * 4);
    fill_f32<<<gdiv((long long)NN * 4, B), B, 0, stream>>>(denom, 0.f, NN * 4);
    fill_f32<<<gdiv((long long)NN * HC, B), B, 0, stream>>>(agg, 0.f, NN * HC);
    edge_max<<<gdiv((long long)ET * 4, B), B, 0, stream>>>(ei, als, ald, emax, 4);
    edge_exp<<<gdiv((long long)ET * 4, B), B, 0, stream>>>(ei, als, ald, emax, pbuf, denom, 4);
    aggregate4<<<gdiv((long long)ET * (HC / 4), B), B, 0, stream>>>(ei, h1, pbuf, denom, agg, 4, 64);
    bias_elu_to_bf16<<<gdiv((long long)NN * HC, B), B, 0, stream>>>(agg, b1, h1b);

    // ---- layer 2: GEMM (WMMA bf16, LDS-staged B), softmax, scatter ----
    wmma_gemm_bf16<<<MBLKS * STRIPS, B, 0, stream>>>(h1b, W2p, h2, NN, 256, 256);
    attn_scores<<<gdiv((long long)NN * 2, B), B, 0, stream>>>(h2, as2, ad2, als, ald, 2, 128);
    fill_f32<<<gdiv((long long)NN * 2, B), B, 0, stream>>>(emax, NEG_INF, NN * 2);
    fill_f32<<<gdiv((long long)NN * 2, B), B, 0, stream>>>(denom, 0.f, NN * 2);
    fill_f32<<<gdiv((long long)NN * HC, B), B, 0, stream>>>(agg, 0.f, NN * HC);
    edge_max<<<gdiv((long long)ET * 2, B), B, 0, stream>>>(ei, als, ald, emax, 2);
    edge_exp<<<gdiv((long long)ET * 2, B), B, 0, stream>>>(ei, als, ald, emax, pbuf, denom, 2);
    aggregate4<<<gdiv((long long)ET * (HC / 4), B), B, 0, stream>>>(ei, h2, pbuf, denom, agg, 2, 128);

    // ---- pooling (fused bias+ELU) + MLP head ----
    fill_f32<<<gdiv((long long)NG * HC, B), B, 0, stream>>>(addp, 0.f, NG * HC);
    fill_f32<<<gdiv((long long)NG * HC, B), B, 0, stream>>>(maxp, NEG_INF, NG * HC);
    fill_f32<<<gdiv((long long)NG, B), B, 0, stream>>>(cnt, 0.f, NG);
    pool_kernel4<<<gdiv((long long)NN * (HC / 4), B), B, 0, stream>>>(agg, b2, batch, addp, maxp, cnt);
    mlp1_kernel<<<gdiv((long long)NG * 64, B), B, 0, stream>>>(addp, maxp, cnt, l1w, l1b, gh);
    mlp2_kernel<<<gdiv((long long)NG, B), B, 0, stream>>>(gh, l2w, l2b, out);
}
